// M3Gnet_51513837748546
// MI455X (gfx1250) — matrix operand order repaired and checked by reading, not compiled, not run on hardware
//
#include <hip/hip_runtime.h>

// ---------------------------------------------------------------------------
// M3GNet forward for MI455X (gfx1250, wave32, WMMA bf16 16x16x32, f32 accum).
//
// Fixed shapes: G=32 graphs, A=128 atoms (N=4096), EG=2500 edges (E=80000),
// TPE=10 triples/edge (T=800000), UNITS=128. All GEMM row counts (E, N) are
// multiples of 16 -> no partial tiles, guards removed from the hot loop.
//
// Assumptions (compile-only environment):
//  * d_in[12:] are params leaves in jax.tree_util order (dict keys sorted).
//  * ws_size >= ~160 MB.
//  * edge->atom scatter uses f32 atomics (standard practice).
// ---------------------------------------------------------------------------

#define G_N    32
#define A_N    128
#define EG     2500
#define TPE    10
#define NATOMS (G_N * A_N)     // 4096
#define NEDGES (G_N * EG)      // 80000
#define NTRIP  (NEDGES * TPE)  // 800000
#define UNITS  128
#define SPH    16

typedef __bf16 bf16;
typedef __attribute__((ext_vector_type(16))) bf16  v16bf;
typedef __attribute__((ext_vector_type(8)))  bf16  v8bf;
typedef __attribute__((ext_vector_type(4)))  bf16  v4bf;
typedef __attribute__((ext_vector_type(8)))  float v8f;

__device__ __constant__ float cSBZ[4][4] = {
    {3.14159265f, 6.28318531f,  9.42477796f, 12.56637061f},
    {4.49340946f, 7.72525184f, 10.90412166f, 14.06619391f},
    {5.76345920f, 9.09501133f, 12.32294097f, 15.51460301f},
    {6.98793200f, 10.41711855f, 13.69802315f, 16.92362129f}};
__device__ __constant__ float cPREF[4] = {0.28209479f, 0.48860251f, 0.63078313f, 0.74635267f};

// Fast native transcendentals: v_exp_f32 + v_rcp_f32 (single-instruction each)
// for gating nonlinearities, v_sin/v_cos for basis functions.
__device__ __forceinline__ float sigm(float x)   { return __builtin_amdgcn_rcpf(1.0f + __expf(-x)); }
__device__ __forceinline__ float swishf(float x) { return x * sigm(x); }
__device__ __forceinline__ float actf(int A, float x) { return A == 1 ? swishf(x) : (A == 2 ? sigm(x) : x); }

__device__ __forceinline__ float sphj(int l, float x) {
    float s = __sinf(x), c = __cosf(x);
    float ix = __builtin_amdgcn_rcpf(x);
    float ix2 = ix * ix, ix3 = ix2 * ix, ix4 = ix2 * ix2, ix5 = ix4 * ix;
    if (l == 0) return s * ix;
    if (l == 1) return s * ix2 - c * ix;
    if (l == 2) return (3.f * ix3 - ix) * s - 3.f * c * ix2;
    if (l == 3) return (15.f * ix4 - 6.f * ix2) * s - (15.f * ix3 - ix) * c;
    return (105.f * ix5 - 45.f * ix3 + ix) * s - (105.f * ix4 - 10.f * ix2) * c;
}

__device__ __forceinline__ float pcut4(float r) {  // poly_cutoff(r, 4.0)
    float x = r * 0.25f;
    float x2 = x * x, x3 = x2 * x;
    return (r <= 4.0f) ? (1.f - 6.f * x3 * x2 + 15.f * x2 * x2 - 10.f * x3) : 0.f;
}

__device__ __forceinline__ float sincx(float x) {
    return (fabsf(x) < 1e-8f) ? 1.f : __sinf(x) * __builtin_amdgcn_rcpf(x);
}

// ---------------------------------------------------------------------------
// Per-edge geometry: evec, |evec|, smooth-bessel RBF (4 values).
// ---------------------------------------------------------------------------
__global__ void k_geom(const float* __restrict__ pos, const float* __restrict__ cell,
                       const float* __restrict__ pbc, const int* __restrict__ eidx,
                       float* __restrict__ evec, float* __restrict__ elen,
                       float* __restrict__ rbf) {
    int e = blockIdx.x * blockDim.x + threadIdx.x;
    if (e >= NEDGES) return;
    int i0 = eidx[e], i1 = eidx[NEDGES + e];
    int g = i0 / A_N;
    float p0 = pbc[e * 3 + 0], p1 = pbc[e * 3 + 1], p2 = pbc[e * 3 + 2];
    const float* C = cell + g * 9;
    float r2 = 0.f;
#pragma unroll
    for (int j = 0; j < 3; j++) {
        float sh = p0 * C[j] + p1 * C[3 + j] + p2 * C[6 + j];
        float d  = pos[i0 * 3 + j] - (pos[i1 * 3 + j] + sh);
        evec[e * 3 + j] = d;
        r2 += d * d;
    }
    float r = sqrtf(r2);
    elen[e] = r;
    // smooth_bessel (MAX_N = 4, CUTOFF = 5)
    const float en1 = 9.f / 65.f, en2 = 64.f / 325.f, en3 = 225.f / 1025.f;
    const float d1 = 1.f - en1, d2 = 1.f - en2 / d1, d3 = 1.f - en3 / d2;
    const float c15 = 11.18033988749895f;          // 5^1.5
    const float s2p = 4.442882938158366f;          // sqrt(2)*pi
    const float k1  = 3.14159265358979f / 5.f;
    float coef0 =  s2p / c15 * 2.f  / sqrtf(5.f);
    float coef1 = -s2p / c15 * 6.f  / sqrtf(13.f);
    float coef2 =  s2p / c15 * 12.f / sqrtf(25.f);
    float coef3 = -s2p / c15 * 20.f / sqrtf(41.f);
    float f0 = coef0 * (sincx(r * 1.f * k1) + sincx(r * 2.f * k1));
    float f1 = coef1 * (sincx(r * 2.f * k1) + sincx(r * 3.f * k1));
    float f2 = coef2 * (sincx(r * 3.f * k1) + sincx(r * 4.f * k1));
    float f3 = coef3 * (sincx(r * 4.f * k1) + sincx(r * 5.f * k1));
    float g0 = f0;
    float g1 = (f1 + sqrtf(en1) * g0) / sqrtf(d1);
    float g2 = (f2 + sqrtf(en2 / d1) * g1) / sqrtf(d2);
    float g3 = (f3 + sqrtf(en3 / d2) * g2) / sqrtf(d3);
    rbf[e * 4 + 0] = g0; rbf[e * 4 + 1] = g1; rbf[e * 4 + 2] = g2; rbf[e * 4 + 3] = g3;
}

// edge_feat = swish(rbf @ edge_enc)   [E,4]@[4,128]
__global__ void k_edge_init(const float* __restrict__ rbf, const float* __restrict__ enc,
                            float* __restrict__ ef) {
    int i = blockIdx.x * blockDim.x + threadIdx.x;
    if (i >= NEDGES * UNITS) return;
    int e = i >> 7, c = i & 127;
    float4 rq = *(const float4*)&rbf[e * 4];
    float s = rq.x * enc[c] + rq.y * enc[128 + c] + rq.z * enc[256 + c] + rq.w * enc[384 + c];
    ef[i] = swishf(s);
}

// Per-triple: sbf (16 values) premultiplied by the two-body cutoffs, plus the
// destination-atom index for the amask gather (= ei1[tbi1]).
__global__ void k_triples(const float* __restrict__ evec, const float* __restrict__ elen,
                          const int* __restrict__ tb, const int* __restrict__ eidx,
                          float* __restrict__ sbfc, int* __restrict__ tbdst) {
    int t = blockIdx.x * blockDim.x + threadIdx.x;
    if (t >= NTRIP) return;
    int g  = t / (EG * TPE);
    int e1 = t / TPE;                  // tbi[:,0] + bias == global edge index
    int e2 = tb[t * 2 + 1] + g * EG;   // tbi[:,1] + bias
    float rij = elen[e1], rik = elen[e2];
    float dd = evec[e1 * 3] * evec[e2 * 3] + evec[e1 * 3 + 1] * evec[e2 * 3 + 1] +
               evec[e1 * 3 + 2] * evec[e2 * 3 + 2];
    float ct = dd / (rij * rik);
    ct = fminf(fmaxf(ct, -1.f + 1e-7f), 1.f - 1e-7f);
    float cut = pcut4(rij) * pcut4(rik);
    float x = fmaxf(rik, 1e-6f) * 0.2f;      // / CUTOFF(=5)
    float P[4];
    P[0] = 1.f; P[1] = ct; P[2] = 0.5f * (3.f * ct * ct - 1.f);
    P[3] = 0.5f * (5.f * ct * ct * ct - 3.f * ct);
#pragma unroll
    for (int l = 0; l < 4; l++) {
        float y = cPREF[l] * P[l];
#pragma unroll
        for (int n = 0; n < 4; n++) {
            float z   = cSBZ[l][n];
            float nrm = 0.1264911064f / fabsf(sphj(l + 1, z));  // sqrt(2/5^3)/|j_{l+1}(z)|
            sbfc[t * 16 + l * 4 + n] = sphj(l, x * z) * nrm * y * cut;
        }
    }
    tbdst[t] = eidx[NEDGES + e2];
}

__global__ void k_atom_init(const int* __restrict__ attr, const float* __restrict__ emb,
                            float* __restrict__ af) {
    int i = blockIdx.x * blockDim.x + threadIdx.x;
    if (i >= NATOMS * UNITS) return;
    int a = i >> 7, c = i & 127;
    af[i] = emb[attr[a] * UNITS + c];
}

// amask = sigmoid(atom_feat @ tb_W + tb_b)   [N,128]@[128,16]  (VALU, tiny)
__global__ void k_amask(const float* __restrict__ af, const float* __restrict__ W,
                        const float* __restrict__ b, float* __restrict__ am) {
    int i = blockIdx.x * blockDim.x + threadIdx.x;
    if (i >= NATOMS * SPH) return;
    int a = i >> 4, c = i & 15;
    float s = b[c];
#pragma unroll 4
    for (int k = 0; k < UNITS; k++) s += af[a * UNITS + k] * W[k * SPH + c];
    am[i] = sigm(s);
}

// new_edge[e] = sum over its 10 contiguous triples of sbf_cut * amask[tb_dst]
// (tbi[:,0] groups 10 consecutive triples per edge -> deterministic, no atomics)
__global__ void k_new_edge(const float* __restrict__ sbfc, const float* __restrict__ am,
                           const int* __restrict__ tbdst, float* __restrict__ ne) {
    int i = blockIdx.x * blockDim.x + threadIdx.x;
    if (i >= NEDGES * SPH) return;
    int e = i >> 4, f = i & 15;
    float s = 0.f;
#pragma unroll
    for (int j = 0; j < TPE; j++) {
        int t = e * TPE + j;
        s += sbfc[t * 16 + f] * am[tbdst[t] * 16 + f];
    }
    ne[i] = s;
}

// ---------------------------------------------------------------------------
// Weight repack: f32 row-major [K x 128] -> bf16 in per-lane B-fragment order
// [colTile(8)][kstep(K/32)][lane(32)][e(16)];  lane: n=lane%16, kh=lane/16,
// fragment element e covers K = kstep*32 + (e&8)*2 + kh*8 + (e&7)  (V_WMMA bf16 B layout).
// ---------------------------------------------------------------------------
__global__ void k_repack(const float* __restrict__ W, bf16* __restrict__ dst,
                         int KS, int realK) {
    int i = blockIdx.x * blockDim.x + threadIdx.x;
    int total = KS * 32 * 16 * 8;
    if (i >= total) return;
    int e = i & 15, lane = (i >> 4) & 31;
    int rem = i >> 9;
    int ks = rem % KS, ct = rem / KS;
    int n = lane & 15, kh = lane >> 4;
    int k = ks * 32 + (e & 8) * 2 + kh * 8 + (e & 7);
    int col = ct * 16 + n;
    float v = (k < realK) ? W[k * 128 + col] : 0.f;
    dst[i] = (bf16)v;
}

// ---------------------------------------------------------------------------
// Fused dual-path WMMA GEMM: v = actV(A@Wv+bv), g = actG(A@Wg+bg), one 16-row
// tile x 128 cols per block iteration (8 waves, one 16x16 WMMA tile per wave
// per path), fused gated epilogue. Row counts are multiples of 16 (no guards).
// A-tile staged in LDS via float4 loads + packed 4x bf16 (8 B) stores.
// ---------------------------------------------------------------------------
enum { AM_F32 = 0, AM_BF16 = 1, AM_CONCAT = 2, AM_PAD = 3 };
enum { EP_STORE = 0, EP_EDGE_PROJ = 1, EP_ATOM_PROJ = 2, EP_EDGE_ADD = 3 };

template <int K, int AMODE, bool DUALA, int ACTV, int ACTG, int EPI>
__launch_bounds__(256, 2)
__global__ void k_gemm_dual(const void* __restrict__ Av, const void* __restrict__ Ag,
                            const float* __restrict__ af, const float* __restrict__ ef,
                            const int* __restrict__ eidx,
                            const bf16* __restrict__ WpV, const float* __restrict__ bV,
                            const bf16* __restrict__ WpG, const float* __restrict__ bG,
                            const float* __restrict__ rbf, const float* __restrict__ lin,
                            bf16* __restrict__ outV, bf16* __restrict__ outG,
                            float* __restrict__ dst, int M, int realK) {
    constexpr int KS = K / 32;
    constexpr int KP = K + 8;    // pad to break LDS bank cycling
    constexpr int KC = K / 4;    // 4-element chunks per row
    __shared__ alignas(32) bf16 AsV[16 * KP];
    __shared__ alignas(32) bf16 AsG[DUALA ? 16 * KP : 16];
    const int lane = threadIdx.x & 31;
    const int wave = threadIdx.x >> 5;  // column tile 0..7
    const int m  = lane & 15;
    const int kh = lane >> 4;
    const int nTiles = M >> 4;          // M is a multiple of 16

    for (int tile = blockIdx.x; tile < nTiles; tile += gridDim.x) {
        const int row0 = tile << 4;
        // cooperative A-tile load, 128-bit global chunks -> packed bf16 in LDS
        for (int idx = threadIdx.x; idx < 16 * KC; idx += 256) {
            int r  = idx / KC;
            int kc = (idx - r * KC) << 2;   // chunk start (multiple of 4)
            int row = row0 + r;
            if (AMODE == AM_BF16) {
                *(v4bf*)&AsV[r * KP + kc] = *(const v4bf*)((const bf16*)Av + (size_t)row * K + kc);
                if (DUALA)
                    *(v4bf*)&AsG[r * KP + kc] = *(const v4bf*)((const bf16*)Ag + (size_t)row * K + kc);
            } else {
                float4 q;
                if (AMODE == AM_F32) {
                    q = *(const float4*)((const float*)Av + (size_t)row * K + kc);
                } else if (AMODE == AM_PAD) {
                    if (kc < realK) q = *(const float4*)((const float*)Av + (size_t)row * realK + kc);
                    else            q = make_float4(0.f, 0.f, 0.f, 0.f);
                } else {  // AM_CONCAT: [ atom_feat[ei0] | atom_feat[ei1] | edge_feat ]
                    const float* sp;
                    if (kc < UNITS)          sp = af + (size_t)eidx[row] * UNITS + kc;
                    else if (kc < 2 * UNITS) sp = af + (size_t)eidx[NEDGES + row] * UNITS + (kc - UNITS);
                    else                     sp = ef + (size_t)row * UNITS + (kc - 2 * UNITS);
                    q = *(const float4*)sp;
                }
                v4bf s;
                s[0] = (bf16)q.x; s[1] = (bf16)q.y; s[2] = (bf16)q.z; s[3] = (bf16)q.w;
                *(v4bf*)&AsV[r * KP + kc] = s;
            }
        }
        __syncthreads();

        v8f accv, accg;
#pragma unroll
        for (int i = 0; i < 8; i++) { accv[i] = 0.f; accg[i] = 0.f; }

#pragma unroll
        for (int ks = 0; ks < KS; ++ks) {
            // A fragment (ISA 16-bit A layout): lane m, elems 0..7 -> K=ks*32+kh*8+i,
            // elems 8..15 -> K=ks*32+16+kh*8+i  => two contiguous 16B LDS reads.
            const bf16* ap = &AsV[m * KP + ks * 32 + kh * 8];
            v8bf alo = *(const v8bf*)ap;
            v8bf ahi = *(const v8bf*)(ap + 16);
            v16bf a = __builtin_shufflevector(alo, ahi, 0, 1, 2, 3, 4, 5, 6, 7,
                                              8, 9, 10, 11, 12, 13, 14, 15);
            v16bf bv = *(const v16bf*)&WpV[((wave * KS + ks) * 32 + lane) * 16];
            accv = __builtin_amdgcn_wmma_f32_16x16x32_bf16(false, a, false, bv,
                                                           (short)0, accv, false, false);
            v16bf ag;
            if (DUALA) {
                const bf16* gp = &AsG[m * KP + ks * 32 + kh * 8];
                v8bf glo = *(const v8bf*)gp;
                v8bf ghi = *(const v8bf*)(gp + 16);
                ag = __builtin_shufflevector(glo, ghi, 0, 1, 2, 3, 4, 5, 6, 7,
                                             8, 9, 10, 11, 12, 13, 14, 15);
            } else {
                ag = a;
            }
            v16bf bg = *(const v16bf*)&WpG[((wave * KS + ks) * 32 + lane) * 16];
            accg = __builtin_amdgcn_wmma_f32_16x16x32_bf16(false, ag, false, bg,
                                                           (short)0, accg, false, false);
        }
        __syncthreads();  // all LDS reads done before next tile overwrites

        // epilogue: C/D layout -> element r is row (row0 + r + 8*kh), col = wave*16+m
        const int col = (wave << 4) + m;
        const float bv_ = bV[col];
        const float bg_ = bG[col];
        float w0 = 0.f, w1 = 0.f, w2 = 0.f, w3 = 0.f;
        if (EPI == EP_EDGE_PROJ || EPI == EP_ATOM_PROJ) {
            w0 = lin[col]; w1 = lin[UNITS + col];
            w2 = lin[2 * UNITS + col]; w3 = lin[3 * UNITS + col];
        }
#pragma unroll
        for (int r = 0; r < 8; r++) {
            int row = row0 + r + (kh << 3);
            float v = actf(ACTV, accv[r] + bv_);
            float g = actf(ACTG, accg[r] + bg_);
            if (EPI == EP_STORE) {
                outV[(size_t)row * UNITS + col] = (bf16)v;
                outG[(size_t)row * UNITS + col] = (bf16)g;
            } else {
                float p = v * g;
                if (EPI == EP_EDGE_PROJ || EPI == EP_ATOM_PROJ) {
                    float4 rq = *(const float4*)&rbf[row * 4];   // one b128 load
                    float pr = rq.x * w0 + rq.y * w1 + rq.z * w2 + rq.w * w3;
                    if (EPI == EP_ATOM_PROJ) pr = swishf(pr);
                    p *= pr;
                }
                if (EPI == EP_ATOM_PROJ) atomicAdd(&dst[(size_t)eidx[row] * UNITS + col], p);
                else                     dst[(size_t)row * UNITS + col] += p;
            }
        }
    }
}

// Final layer (128 -> 1, gated), scale/shift by Z, deterministic per-graph sum.
__global__ void k_final(const bf16* __restrict__ Fv, const bf16* __restrict__ Fg,
                        const float* __restrict__ vW, const float* __restrict__ vb,
                        const float* __restrict__ gW, const float* __restrict__ gb,
                        const int* __restrict__ attr, const float* __restrict__ sc,
                        const float* __restrict__ sh, float* __restrict__ out) {
    __shared__ float red[A_N];
    int g = blockIdx.x, tid = threadIdx.x;
    int a = g * A_N + tid;
    float sv = vb[0], sg = gb[0];
#pragma unroll
    for (int k = 0; k < UNITS; k += 8) {       // 16-byte bf16 chunks
        v8bf qv = *(const v8bf*)&Fv[(size_t)a * UNITS + k];
        v8bf qg = *(const v8bf*)&Fg[(size_t)a * UNITS + k];
#pragma unroll
        for (int j = 0; j < 8; j++) {
            sv += (float)qv[j] * vW[k + j];
            sg += (float)qg[j] * gW[k + j];
        }
    }
    float e = sv * sigm(sg);   // v-path last layer has no activation
    int z = attr[a];
    e = e * sc[z] + sh[z];
    red[tid] = e;
    __syncthreads();
    for (int s = 64; s > 0; s >>= 1) {
        if (tid < s) red[tid] += red[tid + s];
        __syncthreads();
    }
    if (tid == 0) out[g] = red[0];
}

// ---------------------------------------------------------------------------
// Host-side orchestration
// ---------------------------------------------------------------------------
static inline int cdiv(long n, int d) { return (int)((n + d - 1) / d); }

struct BlockP {
    const float *ag0W, *ag0b, *ag1W, *ag1b, *av0W, *av0b, *av1W, *av1b, *atom_lin;
    const float *eg0W, *eg0b, *eg1W, *eg1b, *ev0W, *ev0b, *ev1W, *ev1b, *edge_lin;
    const float *tbW, *tbb, *tgW, *tgb, *tvW, *tvb;
};

extern "C" void kernel_launch(void* const* d_in, const int* in_sizes, int n_in,
                              void* d_out, int out_size, void* d_ws, size_t ws_size,
                              hipStream_t stream) {
    (void)in_sizes; (void)n_in; (void)out_size; (void)ws_size;
    const float* pos  = (const float*)d_in[0];
    const float* cell = (const float*)d_in[1];
    const float* pbc  = (const float*)d_in[2];
    const int*   attr = (const int*)d_in[3];
    const int*   eidx = (const int*)d_in[4];
    const int*   tbi  = (const int*)d_in[5];
    // d_in[6..11]: num_three_body/num_bonds/num_triple_ij/num_atoms/batch/num_graphs
    // are constant for these shapes and baked in.

    int pi = 12;  // params leaves, jax tree order (sorted dict keys)
    auto F = [&]() -> const float* { return (const float*)d_in[pi++]; };
    const float* atom_emb = F();
    BlockP bp[4];
    for (int b = 0; b < 4; b++) {
        bp[b].ag0W = F(); bp[b].ag0b = F(); bp[b].ag1W = F(); bp[b].ag1b = F();
        bp[b].av0W = F(); bp[b].av0b = F(); bp[b].av1W = F(); bp[b].av1b = F();
        bp[b].atom_lin = F();
        bp[b].eg0W = F(); bp[b].eg0b = F(); bp[b].eg1W = F(); bp[b].eg1b = F();
        bp[b].ev0W = F(); bp[b].ev0b = F(); bp[b].ev1W = F(); bp[b].ev1b = F();
        bp[b].edge_lin = F();
        bp[b].tbW = F(); bp[b].tbb = F();
        bp[b].tgW = F(); bp[b].tgb = F();
        bp[b].tvW = F(); bp[b].tvb = F();
    }
    const float* edge_enc = F();
    const float *fg0W = F(), *fg0b = F(), *fg1W = F(), *fg1b = F(), *fg2W = F(), *fg2b = F();
    const float *fv0W = F(), *fv0b = F(), *fv1W = F(), *fv1b = F(), *fv2W = F(), *fv2b = F();
    const float* scaleP = F();
    const float* shiftP = F();

    // workspace carve (256B aligned)
    size_t off = 0;
    auto carve = [&](size_t bytes) -> char* {
        char* p = (char*)d_ws + off;
        off += (bytes + 255) & ~(size_t)255;
        return p;
    };
    float* evecB = (float*)carve((size_t)NEDGES * 3 * 4);
    float* elenB = (float*)carve((size_t)NEDGES * 4);
    float* rbfB  = (float*)carve((size_t)NEDGES * 4 * 4);
    float* sbfc  = (float*)carve((size_t)NTRIP * 16 * 4);
    int*   tbdst = (int*)carve((size_t)NTRIP * 4);
    float* afeat = (float*)carve((size_t)NATOMS * UNITS * 4);
    float* efeat = (float*)carve((size_t)NEDGES * UNITS * 4);
    float* amaskB = (float*)carve((size_t)NATOMS * SPH * 4);
    float* nedge  = (float*)carve((size_t)NEDGES * SPH * 4);
    bf16*  Hv  = (bf16*)carve((size_t)NEDGES * UNITS * 2);
    bf16*  Hg  = (bf16*)carve((size_t)NEDGES * UNITS * 2);
    bf16*  Fv  = (bf16*)carve((size_t)NATOMS * UNITS * 2);
    bf16*  Fg  = (bf16*)carve((size_t)NATOMS * UNITS * 2);
    bf16*  Fv2 = (bf16*)carve((size_t)NATOMS * UNITS * 2);
    bf16*  Fg2 = (bf16*)carve((size_t)NATOMS * UNITS * 2);

    auto packW = [&](const float* src, int Kpad, int realK) -> bf16* {
        bf16* dstp = (bf16*)carve((size_t)Kpad * 128 * 2);
        int total = Kpad * 128;
        k_repack<<<cdiv(total, 256), 256, 0, stream>>>(src, dstp, Kpad / 32, realK);
        return dstp;
    };
    bf16 *Pev0[4], *Peg0[4], *Pev1[4], *Peg1[4], *Pav0[4], *Pag0[4], *Pav1[4], *Pag1[4],
         *Ptv[4], *Ptg[4];
    for (int b = 0; b < 4; b++) {
        Pev0[b] = packW(bp[b].ev0W, 384, 384);
        Peg0[b] = packW(bp[b].eg0W, 384, 384);
        Pev1[b] = packW(bp[b].ev1W, 128, 128);
        Peg1[b] = packW(bp[b].eg1W, 128, 128);
        Pav0[b] = packW(bp[b].av0W, 384, 384);
        Pag0[b] = packW(bp[b].ag0W, 384, 384);
        Pav1[b] = packW(bp[b].av1W, 128, 128);
        Pag1[b] = packW(bp[b].ag1W, 128, 128);
        Ptv[b]  = packW(bp[b].tvW, 32, 16);   // K padded 16 -> 32
        Ptg[b]  = packW(bp[b].tgW, 32, 16);
    }
    bf16* Pfv0 = packW(fv0W, 128, 128);
    bf16* Pfg0 = packW(fg0W, 128, 128);
    bf16* Pfv1 = packW(fv1W, 128, 128);
    bf16* Pfg1 = packW(fg1W, 128, 128);

    // ---- featurization ----
    k_geom<<<cdiv(NEDGES, 256), 256, 0, stream>>>(pos, cell, pbc, eidx, evecB, elenB, rbfB);
    k_edge_init<<<cdiv((long)NEDGES * UNITS, 256), 256, 0, stream>>>(rbfB, edge_enc, efeat);
    k_triples<<<cdiv(NTRIP, 256), 256, 0, stream>>>(evecB, elenB, tbi, eidx, sbfc, tbdst);
    k_atom_init<<<cdiv((long)NATOMS * UNITS, 256), 256, 0, stream>>>(attr, atom_emb, afeat);

    const int GB = 2048;  // grid-stride blocks for E-row GEMMs (5000 tiles)
    for (int b = 0; b < 4; b++) {
        const BlockP& B = bp[b];
        k_amask<<<cdiv((long)NATOMS * SPH, 256), 256, 0, stream>>>(afeat, B.tbW, B.tbb, amaskB);
        k_new_edge<<<cdiv((long)NEDGES * SPH, 256), 256, 0, stream>>>(sbfc, amaskB, tbdst, nedge);
        // edge_feat += swish(ne@Wv+bv) * sigmoid(ne@Wg+bg)
        k_gemm_dual<32, AM_PAD, false, 1, 2, EP_EDGE_ADD><<<GB, 256, 0, stream>>>(
            nedge, nullptr, nullptr, nullptr, nullptr,
            Ptv[b], B.tvb, Ptg[b], B.tgb, nullptr, nullptr,
            nullptr, nullptr, efeat, NEDGES, 16);
        // edge gmlp layer 0 (concat input, both paths swish) -> Hv, Hg (bf16)
        k_gemm_dual<384, AM_CONCAT, false, 1, 1, EP_STORE><<<GB, 256, 0, stream>>>(
            nullptr, nullptr, afeat, efeat, eidx,
            Pev0[b], B.ev0b, Peg0[b], B.eg0b, nullptr, nullptr,
            Hv, Hg, nullptr, NEDGES, 384);
        // edge gmlp layer 1 + gated epilogue: edge_feat += v*g*(rbf@edge_lin)
        k_gemm_dual<128, AM_BF16, true, 1, 2, EP_EDGE_PROJ><<<GB, 256, 0, stream>>>(
            Hv, Hg, nullptr, nullptr, eidx,
            Pev1[b], B.ev1b, Peg1[b], B.eg1b, rbfB, B.edge_lin,
            nullptr, nullptr, efeat, NEDGES, 128);
        // atom gmlp layer 0 (uses updated edge_feat)
        k_gemm_dual<384, AM_CONCAT, false, 1, 1, EP_STORE><<<GB, 256, 0, stream>>>(
            nullptr, nullptr, afeat, efeat, eidx,
            Pav0[b], B.av0b, Pag0[b], B.ag0b, nullptr, nullptr,
            Hv, Hg, nullptr, NEDGES, 384);
        // atom gmlp layer 1 + scatter: atom_feat[ei0] += v*g*swish(rbf@atom_lin)
        k_gemm_dual<128, AM_BF16, true, 1, 2, EP_ATOM_PROJ><<<GB, 256, 0, stream>>>(
            Hv, Hg, nullptr, nullptr, eidx,
            Pav1[b], B.av1b, Pag1[b], B.ag1b, rbfB, B.atom_lin,
            nullptr, nullptr, afeat, NEDGES, 128);
    }

    // final gated MLP: two dual layers then 128->1 + per-graph reduction
    k_gemm_dual<128, AM_F32, false, 1, 1, EP_STORE><<<256, 256, 0, stream>>>(
        afeat, nullptr, nullptr, nullptr, nullptr,
        Pfv0, fv0b, Pfg0, fg0b, nullptr, nullptr, Fv, Fg, nullptr, NATOMS, 128);
    k_gemm_dual<128, AM_BF16, true, 1, 1, EP_STORE><<<256, 256, 0, stream>>>(
        Fv, Fg, nullptr, nullptr, nullptr,
        Pfv1, fv1b, Pfg1, fg1b, nullptr, nullptr, Fv2, Fg2, nullptr, NATOMS, 128);
    k_final<<<G_N, A_N, 0, stream>>>(Fv2, Fg2, fv2W, fv2b, fg2W, fg2b,
                                     attr, scaleP, shiftP, (float*)d_out);
}
